// CausalAttention_50457275794110
// MI455X (gfx1250) — compile-verified, matrix operand order checked by reading, not run on hardware
//
#include <hip/hip_runtime.h>

typedef _Float16 h16;
typedef __attribute__((ext_vector_type(16))) _Float16 v16h;
typedef __attribute__((ext_vector_type(8)))  float    v8f;

#define DM     1024
#define NH     16
#define DH     64
#define SEQ_   2048
#define BATCH_ 2

union FragA { v16h h; unsigned int u[8]; };
union PairH { unsigned int u; h16 h[2]; };

__device__ __forceinline__ void lds_fence() {
  asm volatile("s_wait_dscnt 0" ::: "memory");
}
__device__ __forceinline__ void async_fence() {
  asm volatile("s_wait_asynccnt 0" ::: "memory");
}
// one lane's 16B global->LDS copy through the async path (no VGPR data)
__device__ __forceinline__ void async_copy16(unsigned lds_off, const h16* g) {
  unsigned long long ga = (unsigned long long)(uintptr_t)g;
  asm volatile("global_load_async_to_lds_b128 %0, %1, off"
               :: "v"(lds_off), "v"(ga) : "memory");
}

// ---------------------------------------------------------------- fp32 -> f16
__global__ void cvt_f32_f16(const float* __restrict__ s, h16* __restrict__ d, int n) {
  int i = blockIdx.x * blockDim.x + threadIdx.x;
  int st = gridDim.x * blockDim.x;
  for (; i < n; i += st) d[i] = (h16)s[i];
}

// ------------------------------------------------- LDS-tiled WMMA GEMM + bias
// Y[M][N] = A[M][K] * W[K][N] + bias ; block tile 128x64, 4 waves (128 thr),
// each wave computes a 32x64 strip (2x4 wmma accumulators).
template <bool F32OUT>
__global__ __launch_bounds__(128) void gemm_bias_wmma(
    const h16* __restrict__ A, const h16* __restrict__ W,
    const float* __restrict__ bias, void* __restrict__ Yv,
    int M, int K, int N) {
  __shared__ h16 sA[128][40];  // row stride 80B: 16B-aligned rows for async B128
  __shared__ h16 sWt[64][34];  // [n][k] (W staged transposed), 17-dword stride

  const int tid  = threadIdx.x;
  const int lane = tid & 31;
  const int wv   = tid >> 5;
  const int mloc = lane & 15;
  const int hi   = lane >> 4;
  const int rbase = blockIdx.y * 128;
  const int nbase = blockIdx.x * 64;

  v8f zero = {};
  v8f acc[2][4];
#pragma unroll
  for (int mt = 0; mt < 2; ++mt)
#pragma unroll
    for (int nt = 0; nt < 4; ++nt) acc[mt][nt] = zero;

  for (int kc = 0; kc < K; kc += 32) {
    // ---- stage A tile 128x32 via async global->LDS (512 x 16B chunks)
#pragma unroll
    for (int it = 0; it < 4; ++it) {
      int u = it * 128 + tid;
      int row = u >> 2, seg = u & 3;
      unsigned lds_off = (unsigned)(uintptr_t)&sA[row][seg * 8];
      async_copy16(lds_off, A + (size_t)(rbase + row) * K + kc + seg * 8);
    }
    // ---- stage W tile 32x64, transposed into sWt[n][k]
#pragma unroll
    for (int it = 0; it < 2; ++it) {
      int u = it * 128 + tid;
      int kr = u >> 3, seg = u & 7;
      uint4 t = *(const uint4*)(W + (size_t)(kc + kr) * N + nbase + seg * 8);
      alignas(16) h16 tmp[8];
      *(uint4*)tmp = t;
#pragma unroll
      for (int j = 0; j < 8; ++j) sWt[seg * 8 + j][kr] = tmp[j];
    }
    if (kc + 32 < K) {  // prefetch next K-chunk into cache
      __builtin_prefetch(A + (size_t)(rbase + (tid >> 2)) * K + kc + 32, 0, 1);
      __builtin_prefetch(W + (size_t)(kc + 32 + (tid >> 3)) * N + nbase, 0, 1);
    }
    async_fence();
    __syncthreads();

    FragA af[2];
#pragma unroll
    for (int mt = 0; mt < 2; ++mt)
#pragma unroll
      for (int r = 0; r < 8; ++r) {
        int koff = ((r & 4) << 2) + hi * 8 + ((r & 3) << 1);
        af[mt].u[r] = *(const unsigned int*)&sA[wv * 32 + mt * 16 + mloc][koff];
      }
#pragma unroll
    for (int nt = 0; nt < 4; ++nt) {
      FragA bf;
#pragma unroll
      for (int r = 0; r < 8; ++r) {
        int koff = ((r & 4) << 2) + hi * 8 + ((r & 3) << 1);
        bf.u[r] = *(const unsigned int*)&sWt[nt * 16 + mloc][koff];
      }
#pragma unroll
      for (int mt = 0; mt < 2; ++mt)
        acc[mt][nt] = __builtin_amdgcn_wmma_f32_16x16x32_f16(
            false, af[mt].h, false, bf.h, (short)0, acc[mt][nt], false, false);
    }
    __syncthreads();
  }

  // ---- epilogue: bias + store (C/D layout: reg r -> row r+8*hi, lane&15 -> col)
#pragma unroll
  for (int mt = 0; mt < 2; ++mt)
#pragma unroll
    for (int nt = 0; nt < 4; ++nt) {
      int col = nbase + nt * 16 + mloc;
      float bv = bias[col];
#pragma unroll
      for (int r = 0; r < 8; ++r) {
        int row = rbase + wv * 32 + mt * 16 + r + hi * 8;
        float val = acc[mt][nt][r] + bv;
        if (F32OUT)
          ((float*)Yv)[(size_t)row * N + col] = val;
        else
          ((h16*)Yv)[(size_t)row * N + col] = (h16)val;
      }
    }
}

// ----------------------------------------------- flash attention, 1 wave/blk
// grid: (SEQ/16, BATCH*NH), block 32. Q/K/V/H stored [B,S,D] f16, head at
// column offset h*64, so row stride is DM.
__global__ __launch_bounds__(32) void attn_wmma(
    const h16* __restrict__ Qh, const h16* __restrict__ Kh,
    const h16* __restrict__ Vh, h16* __restrict__ Hh) {
  __shared__ h16 sP[16][34];
  const int lane = threadIdx.x & 31;
  const int mloc = lane & 15;
  const int hi   = lane >> 4;
  const int bh   = blockIdx.y;
  const int b    = bh >> 4;   // NH == 16
  const int h    = bh & 15;
  const int qbase = blockIdx.x * 16;

  const size_t headoff = (size_t)b * SEQ_ * DM + (size_t)h * DH;
  const h16* Qp = Qh + headoff;
  const h16* Kp = Kh + headoff;
  const h16* Vp = Vh + headoff;
  h16* Hp = Hh + headoff;

  // Q A-fragments (2 chunks of K=32 over Dh=64), pre-scaled by 1/sqrt(64)
  FragA qa[2];
#pragma unroll
  for (int c = 0; c < 2; ++c) {
#pragma unroll
    for (int r = 0; r < 8; ++r) {
      int koff = ((r & 4) << 2) + hi * 8 + ((r & 3) << 1);
      PairH t;
      t.u = *(const unsigned int*)(Qp + (size_t)(qbase + mloc) * DM + c * 32 + koff);
      t.h[0] = t.h[0] * (h16)0.125f;   // exact (power of 2)
      t.h[1] = t.h[1] * (h16)0.125f;
      qa[c].u[r] = t.u;
    }
  }

  float mrow[8], lrow[8];
#pragma unroll
  for (int r = 0; r < 8; ++r) { mrow[r] = -1e30f; lrow[r] = 0.0f; }
  v8f zero = {};
  v8f O0 = zero, O1 = zero, O2 = zero, O3 = zero;

  for (int kb = 0; kb <= qbase + 15; kb += 32) {   // causal: only lower blocks
    // ---- scores S[16q][32k] = Q*K^T  (B frag lane n = key column)
    v8f s0 = zero, s1 = zero;
#pragma unroll
    for (int c = 0; c < 2; ++c) {
      FragA k0, k1;
#pragma unroll
      for (int r = 0; r < 8; ++r) {
        int koff = ((r & 4) << 2) + hi * 8 + ((r & 3) << 1);
        int dh = c * 32 + koff;
        k0.u[r] = *(const unsigned int*)(Kp + (size_t)(kb + mloc) * DM + dh);
        k1.u[r] = *(const unsigned int*)(Kp + (size_t)(kb + 16 + mloc) * DM + dh);
      }
      s0 = __builtin_amdgcn_wmma_f32_16x16x32_f16(false, qa[c].h, false, k0.h,
                                                  (short)0, s0, false, false);
      s1 = __builtin_amdgcn_wmma_f32_16x16x32_f16(false, qa[c].h, false, k1.h,
                                                  (short)0, s1, false, false);
    }

    // ---- causal mask + online softmax (row = reg r within a 16-lane half)
    const int key0 = kb + mloc;
    const int key1 = kb + 16 + mloc;
    lds_fence();  // previous iteration's sP reads complete before overwrite
#pragma unroll
    for (int r = 0; r < 8; ++r) {
      int q = qbase + r + hi * 8;
      float a0 = (key0 > q) ? -1e30f : s0[r];
      float a1 = (key1 > q) ? -1e30f : s1[r];
      float t = fmaxf(a0, a1);
      t = fmaxf(t, __shfl_xor(t, 1));
      t = fmaxf(t, __shfl_xor(t, 2));
      t = fmaxf(t, __shfl_xor(t, 4));
      t = fmaxf(t, __shfl_xor(t, 8));
      float mnew  = fmaxf(mrow[r], t);
      float alpha = __expf(mrow[r] - mnew);
      mrow[r] = mnew;
      float p0 = __expf(a0 - mnew);
      float p1 = __expf(a1 - mnew);
      float rs = p0 + p1;
      rs += __shfl_xor(rs, 1);
      rs += __shfl_xor(rs, 2);
      rs += __shfl_xor(rs, 4);
      rs += __shfl_xor(rs, 8);
      lrow[r] = alpha * lrow[r] + rs;
      O0[r] *= alpha; O1[r] *= alpha; O2[r] *= alpha; O3[r] *= alpha;
      sP[r + hi * 8][mloc]      = (h16)p0;   // re-shape P via LDS
      sP[r + hi * 8][16 + mloc] = (h16)p1;
    }
    lds_fence();

    FragA pa;
#pragma unroll
    for (int r = 0; r < 8; ++r) {
      int koff = ((r & 4) << 2) + hi * 8 + ((r & 3) << 1);
      pa.u[r] = *(const unsigned int*)&sP[mloc][koff];
    }

    // ---- O += P(16x32) * V(32x64)
#pragma unroll
    for (int t4 = 0; t4 < 4; ++t4) {
      FragA vb;
#pragma unroll
      for (int r = 0; r < 8; ++r) {
        int koff = ((r & 4) << 2) + hi * 8 + ((r & 3) << 1);
        int dh = t4 * 16 + mloc;
        PairH w;
        w.h[0] = Vp[(size_t)(kb + koff) * DM + dh];
        w.h[1] = Vp[(size_t)(kb + koff + 1) * DM + dh];
        vb.u[r] = w.u;
      }
      v8f& O = t4 == 0 ? O0 : t4 == 1 ? O1 : t4 == 2 ? O2 : O3;
      O = __builtin_amdgcn_wmma_f32_16x16x32_f16(false, pa.h, false, vb.h,
                                                 (short)0, O, false, false);
    }
  }

  // ---- H = O / l
#pragma unroll
  for (int t4 = 0; t4 < 4; ++t4) {
    v8f O = t4 == 0 ? O0 : t4 == 1 ? O1 : t4 == 2 ? O2 : O3;
    int dh = t4 * 16 + mloc;
#pragma unroll
    for (int r = 0; r < 8; ++r) {
      int row = qbase + r + hi * 8;
      Hp[(size_t)row * DM + dh] = (h16)(O[r] / lrow[r]);
    }
  }
}

// --------------------------------------------------------------------- launch
extern "C" void kernel_launch(void* const* d_in, const int* in_sizes, int n_in,
                              void* d_out, int out_size, void* d_ws, size_t ws_size,
                              hipStream_t stream) {
  (void)in_sizes; (void)n_in; (void)out_size; (void)ws_size;
  const float* x  = (const float*)d_in[0];
  const float* Wq = (const float*)d_in[1];
  const float* bq = (const float*)d_in[2];
  const float* Wk = (const float*)d_in[3];
  const float* bk = (const float*)d_in[4];
  const float* Wv = (const float*)d_in[5];
  const float* bv = (const float*)d_in[6];
  const float* Wo = (const float*)d_in[7];
  const float* bo = (const float*)d_in[8];

  const size_t XE = (size_t)BATCH_ * SEQ_ * DM;  // 4Mi elems
  const size_t WE = (size_t)DM * DM;             // 1Mi elems
  h16* xh  = (h16*)d_ws;       // ws usage: 24Mi f16 = 48 MiB
  h16* wqh = xh + XE;
  h16* wkh = wqh + WE;
  h16* wvh = wkh + WE;
  h16* woh = wvh + WE;
  h16* qh  = woh + WE;
  h16* kh  = qh + XE;
  h16* vh  = kh + XE;
  h16* hh  = vh + XE;

  cvt_f32_f16<<<dim3(1024), dim3(256), 0, stream>>>(x, xh, (int)XE);
  cvt_f32_f16<<<dim3(512), dim3(256), 0, stream>>>(Wq, wqh, (int)WE);
  cvt_f32_f16<<<dim3(512), dim3(256), 0, stream>>>(Wk, wkh, (int)WE);
  cvt_f32_f16<<<dim3(512), dim3(256), 0, stream>>>(Wv, wvh, (int)WE);
  cvt_f32_f16<<<dim3(512), dim3(256), 0, stream>>>(Wo, woh, (int)WE);

  const int Mtot = BATCH_ * SEQ_;
  dim3 ggrid(DM / 64, Mtot / 128);  // 16 x 32
  gemm_bias_wmma<false><<<ggrid, 128, 0, stream>>>(xh, wqh, bq, (void*)qh, Mtot, DM, DM);
  gemm_bias_wmma<false><<<ggrid, 128, 0, stream>>>(xh, wkh, bk, (void*)kh, Mtot, DM, DM);
  gemm_bias_wmma<false><<<ggrid, 128, 0, stream>>>(xh, wvh, bv, (void*)vh, Mtot, DM, DM);

  attn_wmma<<<dim3(SEQ_ / 16, BATCH_ * NH), 32, 0, stream>>>(qh, kh, vh, hh);

  gemm_bias_wmma<true><<<ggrid, 128, 0, stream>>>(hh, woh, bo, d_out, Mtot, DM, DM);
}
